// WindowAttention_46377056862447
// MI455X (gfx1250) — matrix operand order repaired and checked by reading, not compiled, not run on hardware
//
#include <hip/hip_runtime.h>

#define NTOK   64
#define HEADS  16
#define DIM    1024
#define HD     64
#define TAB    15
#define BATCH  8
#define NWIN   64
#define BTOT   512

typedef __attribute__((ext_vector_type(16))) __bf16 v16bf;
typedef __attribute__((ext_vector_type(8)))  float  v8f;
typedef __attribute__((__vector_size__(16))) int    vi4;   // int4 for async-LDS builtin

union FragU { v16bf v; uint4 u[2]; };

__device__ __forceinline__ unsigned short f2bf(float f) {
    unsigned u = __float_as_uint(f);
    u += 0x7FFFu + ((u >> 16) & 1u);          // round-to-nearest-even
    return (unsigned short)(u >> 16);
}

// A fragment: lane holds output-row r; K chunks at kh*8 and kh*8+16 (ISA 16-bit A 16x32 layout)
__device__ __forceinline__ v16bf load_fragA(const unsigned short* rowp, int kh) {
    FragU f;
    f.u[0] = *(const uint4*)(rowp + kh * 8);
    f.u[1] = *(const uint4*)(rowp + kh * 8 + 16);
    return f.v;
}
// B fragment: lane holds output-column n (row of B^T); K chunks at kh*8 and kh*8+16 (matches A layout)
__device__ __forceinline__ v16bf load_fragB(const unsigned short* rowp, int kh) {
    FragU f;
    f.u[0] = *(const uint4*)(rowp + kh * 8);
    f.u[1] = *(const uint4*)(rowp + kh * 8 + 16);
    return f.v;
}

// ---------------- fp32 -> bf16 bulk convert ----------------
__global__ void k_convert(const float* __restrict__ src, unsigned short* __restrict__ dst, int n) {
    int i = blockIdx.x * blockDim.x + threadIdx.x;
    int stride = gridDim.x * blockDim.x;
    for (; i < n; i += stride) dst[i] = f2bf(src[i]);
}

// ---------------- fused positional bias: A_phi + A_theta per (img, head) ----------------
__global__ void k_bias(const float* __restrict__ theta_max,
                       const float* __restrict__ a_p, const float* __restrict__ b_p,
                       const float* __restrict__ a_r, const float* __restrict__ b_r,
                       float* __restrict__ bias) {
    int blk = blockIdx.x;                 // img*HEADS + h
    int img = blk >> 4, h = blk & 15;
    float tmax = theta_max[img];
    for (int f = threadIdx.x; f < NTOK * NTOK; f += blockDim.x) {
        int i = f >> 6, j = f & 63;
        int rad = (i >> 3) - (j >> 3);
        int az  = (i & 7)  - (j & 7);
        int ridx = rad < 0 ? rad + TAB : rad;   // python mod TAB
        int aidx = az  < 0 ? az  + TAB : az;
        float az_ang = (float)az * 0.09817477042468103f;  // 2*pi/64
        float rn     = (float)rad * tmax * (1.0f / 64.0f);
        float aphi = a_p[aidx * HEADS + h] * __cosf(az_ang) + b_p[aidx * HEADS + h] * __sinf(az_ang);
        float ath  = a_r[ridx * HEADS + h] * __cosf(rn)     + b_r[ridx * HEADS + h] * __sinf(rn);
        bias[(size_t)blk * (NTOK * NTOK) + f] = aphi + ath;
    }
}

// ---------------- per-(window, head) fused QKV + attention ----------------
__global__ __launch_bounds__(256) void k_qkv_attn(
    const unsigned short* __restrict__ xb,   // (512,64,1024) bf16
    const unsigned short* __restrict__ qw,   // (3072,1024)  bf16
    const float*          __restrict__ qb,   // (3072,)
    const float*          __restrict__ bias, // (8,16,64,64)
    unsigned short*       __restrict__ ao)   // (512,64,1024) bf16
{
    __shared__ __align__(16) unsigned short q_s [NTOK * HD];
    __shared__ __align__(16) unsigned short k_s [NTOK * HD];
    __shared__ __align__(16) unsigned short vT_s[HD * NTOK];
    __shared__ __align__(16) float          sb_s[NTOK * NTOK]; // bias, then scores (same owner thread)
    __shared__ __align__(16) unsigned short p_s [NTOK * NTOK];

    const int bx  = blockIdx.x;
    const int win = bx >> 4, h = bx & 15, img = win >> 6;
    const int tid = threadIdx.x;
    // wave index is wave-uniform: force it scalar so tile selection uses SGPRs / s_cbranch
    const int wave = __builtin_amdgcn_readfirstlane(tid >> 5);
    const int lane = tid & 31;                        // wave32
    const int r = lane & 15, kh = lane >> 4;

    // ---- stage bias tile into LDS; overlap with the QKV GEMM below ----
    const float* bsrc = bias + (size_t)(img * HEADS + h) * (NTOK * NTOK);
#if defined(__has_builtin) && __has_builtin(__builtin_amdgcn_global_load_async_to_lds_b128)
    // gfx1250 async copy: tracked by ASYNCcnt, no VGPR round-trip
    for (int f = tid * 4; f < NTOK * NTOK; f += 256 * 4)
        __builtin_amdgcn_global_load_async_to_lds_b128(
            (__attribute__((address_space(1))) vi4*)(bsrc + f),
            (__attribute__((address_space(3))) vi4*)(sb_s + f), 0, 0);
#else
    for (int f = tid; f < NTOK * NTOK; f += 256)
        sb_s[f] = bsrc[f];
#endif

    // ---- QKV GEMM: 64 rows x 192 cols (q|k|v slices of this head), K=1024 ----
    const int mt  = wave & 3;            // 16-row block
    const int ntb = (wave >> 2) * 6;     // 6 of 12 16-col tiles
    v8f acc[6] = {};

    const unsigned short* arow = xb + ((size_t)win * NTOK + mt * 16 + r) * DIM;
    #pragma unroll 1
    for (int ks = 0; ks < DIM / 32; ++ks) {
        __builtin_prefetch(arow + (ks + 4) * 32, 0, 0);   // global_prefetch_b8 (stays inside ws)
        v16bf af = load_fragA(arow + ks * 32, kh);
        #pragma unroll
        for (int t = 0; t < 6; ++t) {
            int nt   = ntb + t;
            int col0 = (nt >> 2) * (HEADS * HD) + h * HD + (nt & 3) * 16;   // q/k/v section
            v16bf bf = load_fragB(qw + (size_t)(col0 + r) * DIM + ks * 32, kh);
            acc[t] = __builtin_amdgcn_wmma_f32_16x16x32_bf16(false, af, false, bf,
                                                             (short)0, acc[t], false, false);
        }
    }

    // epilogue: +bias, scale q, write q/k to LDS, v transposed (sec is SGPR-uniform now)
    #pragma unroll
    for (int t = 0; t < 6; ++t) {
        int nt   = ntb + t;
        int sec  = nt >> 2;
        int nloc = (nt & 3) * 16 + r;    // column within head (0..63)
        float bv = qb[sec * (HEADS * HD) + h * HD + nloc];
        #pragma unroll
        for (int g = 0; g < 8; ++g) {
            int m = mt * 16 + g + 8 * kh;
            float v = acc[t][g] + bv;
            if      (sec == 0) q_s [m * HD + nloc]    = f2bf(v * 0.125f);  // SCALE = 64^-0.5
            else if (sec == 1) k_s [m * HD + nloc]    = f2bf(v);
            else               vT_s[nloc * NTOK + m]  = f2bf(v);
        }
    }
#if defined(__has_builtin) && __has_builtin(__builtin_amdgcn_global_load_async_to_lds_b128)
  #if defined(__has_builtin) && __has_builtin(__builtin_amdgcn_s_wait_asynccnt)
    __builtin_amdgcn_s_wait_asynccnt(0);
  #else
    asm volatile("s_wait_asynccnt 0x0" ::: "memory");
  #endif
#endif
    __syncthreads();

    // ---- S = q @ k^T  (+bias), 16 tiles of 16x16, K=64 ----
    v8f sacc[2] = {};
    #pragma unroll
    for (int tt = 0; tt < 2; ++tt) {
        int t = wave + tt * 8;
        int mt2 = t & 3, nt2 = t >> 2;
        #pragma unroll
        for (int ks = 0; ks < 2; ++ks) {
            v16bf af = load_fragA(q_s + (mt2 * 16 + r) * HD + ks * 32, kh);
            v16bf bf = load_fragB(k_s + (nt2 * 16 + r) * HD + ks * 32, kh);
            sacc[tt] = __builtin_amdgcn_wmma_f32_16x16x32_bf16(false, af, false, bf,
                                                               (short)0, sacc[tt], false, false);
        }
    }
    #pragma unroll
    for (int tt = 0; tt < 2; ++tt) {
        int t = wave + tt * 8;
        int mt2 = t & 3, nt2 = t >> 2;
        #pragma unroll
        for (int g = 0; g < 8; ++g) {
            int i = mt2 * 16 + g + 8 * kh, j = nt2 * 16 + r;
            sb_s[i * NTOK + j] = sacc[tt][g] + sb_s[i * NTOK + j];  // same-thread RMW, no hazard
        }
    }
    __syncthreads();

    // ---- softmax (one row per thread) -> bf16 P ----
    if (tid < NTOK) {
        float row[NTOK];
        float mx = -1e30f;
        #pragma unroll
        for (int j = 0; j < NTOK; ++j) { row[j] = sb_s[tid * NTOK + j]; mx = fmaxf(mx, row[j]); }
        float sum = 0.f;
        #pragma unroll
        for (int j = 0; j < NTOK; ++j) { row[j] = __expf(row[j] - mx); sum += row[j]; }
        float inv = 1.0f / sum;
        #pragma unroll
        for (int j = 0; j < NTOK; ++j) p_s[tid * NTOK + j] = f2bf(row[j] * inv);
    }
    __syncthreads();

    // ---- O = P @ V  (B-fragments read rows of V^T = columns of V) ----
    v8f oacc[2] = {};
    #pragma unroll
    for (int tt = 0; tt < 2; ++tt) {
        int t = wave + tt * 8;
        int mt2 = t & 3, nt2 = t >> 2;
        #pragma unroll
        for (int ks = 0; ks < 2; ++ks) {
            v16bf af = load_fragA(p_s  + (mt2 * 16 + r) * NTOK + ks * 32, kh);
            v16bf bf = load_fragB(vT_s + (nt2 * 16 + r) * NTOK + ks * 32, kh);
            oacc[tt] = __builtin_amdgcn_wmma_f32_16x16x32_bf16(false, af, false, bf,
                                                               (short)0, oacc[tt], false, false);
        }
    }
    #pragma unroll
    for (int tt = 0; tt < 2; ++tt) {
        int t = wave + tt * 8;
        int mt2 = t & 3, nt2 = t >> 2;
        #pragma unroll
        for (int g = 0; g < 8; ++g) {
            int i = mt2 * 16 + g + 8 * kh;   // token
            int d = nt2 * 16 + r;            // head dim
            ao[((size_t)win * NTOK + i) * DIM + h * HD + d] = f2bf(oacc[tt][g]);
        }
    }
}

// ---------------- output projection: (32768,1024) @ proj_w^T + b -> fp32 ----------------
__global__ __launch_bounds__(256) void k_proj(
    const unsigned short* __restrict__ ao,   // (32768,1024) bf16
    const unsigned short* __restrict__ pw,   // (1024,1024)  bf16
    const float*          __restrict__ pb,
    float*                __restrict__ out)
{
    const int bx   = blockIdx.x;
    const int mblk = bx >> 2;                 // rows mblk*64
    const int nblk = bx & 3;                  // cols nblk*256
    const int tid  = threadIdx.x;
    const int wave = __builtin_amdgcn_readfirstlane(tid >> 5);
    const int lane = tid & 31;
    const int r = lane & 15, kh = lane >> 4;
    const int mt  = wave & 3;
    const int nt0 = (wave >> 2) * 8;          // 8 of 16 col tiles

    v8f acc[8] = {};
    const unsigned short* arow = ao + ((size_t)mblk * 64 + mt * 16 + r) * DIM;
    #pragma unroll 1
    for (int ks = 0; ks < DIM / 32; ++ks) {
        __builtin_prefetch(arow + (ks + 4) * 32, 0, 0);   // global_prefetch_b8 (stays inside ws)
        v16bf af = load_fragA(arow + ks * 32, kh);
        #pragma unroll
        for (int t = 0; t < 8; ++t) {
            int col = nblk * 256 + (nt0 + t) * 16;
            v16bf bf = load_fragB(pw + (size_t)(col + r) * DIM + ks * 32, kh);
            acc[t] = __builtin_amdgcn_wmma_f32_16x16x32_bf16(false, af, false, bf,
                                                             (short)0, acc[t], false, false);
        }
    }
    #pragma unroll
    for (int t = 0; t < 8; ++t) {
        int col = nblk * 256 + (nt0 + t) * 16 + r;
        float bv = pb[col];
        #pragma unroll
        for (int g = 0; g < 8; ++g) {
            int row = mblk * 64 + mt * 16 + g + 8 * kh;
            out[(size_t)row * DIM + col] = acc[t][g] + bv;
        }
    }
}

extern "C" void kernel_launch(void* const* d_in, const int* in_sizes, int n_in,
                              void* d_out, int out_size, void* d_ws, size_t ws_size,
                              hipStream_t stream) {
    const float* x  = (const float*)d_in[0];
    const float* tm = (const float*)d_in[1];
    const float* qw = (const float*)d_in[2];
    const float* qb = (const float*)d_in[3];
    const float* pw = (const float*)d_in[4];
    const float* pb = (const float*)d_in[5];
    const float* ap = (const float*)d_in[6];
    const float* bp = (const float*)d_in[7];
    const float* ar = (const float*)d_in[8];
    const float* br = (const float*)d_in[9];

    char* ws = (char*)d_ws;
    float*          bias = (float*)ws;                                        //  2 MB
    unsigned short* xb   = (unsigned short*)(ws + (size_t)(2u  << 20));       // 64 MB
    unsigned short* qwb  = (unsigned short*)(ws + (size_t)(66u << 20));       //  6 MB
    unsigned short* pwb  = (unsigned short*)(ws + (size_t)(72u << 20));       //  2 MB
    unsigned short* ao   = (unsigned short*)(ws + (size_t)(74u << 20));       // 64 MB

    k_convert<<<4096, 256, 0, stream>>>(x,  xb,  BTOT * NTOK * DIM);
    k_convert<<<1024, 256, 0, stream>>>(qw, qwb, 3 * DIM * DIM);
    k_convert<<<512,  256, 0, stream>>>(pw, pwb, DIM * DIM);
    k_bias   <<<BATCH * HEADS, 256, 0, stream>>>(tm, ap, bp, ar, br, bias);
    k_qkv_attn<<<BTOT * HEADS, 256, 0, stream>>>(xb, qwb, qb, bias, ao);
    k_proj   <<<(BTOT * NTOK / 64) * (DIM / 256), 256, 0, stream>>>(ao, pwb, pb, (float*)d_out);
}
